// CrossAttention_31834297598820
// MI455X (gfx1250) — matrix operand order repaired and checked by reading, not compiled, not run on hardware
//
#include <hip/hip_runtime.h>
#include <math.h>

// ---------------------------------------------------------------------------
// CrossAttention forward for MI455X (gfx1250, wave32, WMMA + TDM).
//   y = softmax((q Wq)(p Wk)^T / sqrt(D)) (p Wv) Wo
// B=2, T=S=2048, C=1024, H=16, D=64.  ~69 GFLOP, ~66MB I/O -> compute bound.
// f16 operands + v_wmma_f32_16x16x32_f16 (fp32 accumulate); flash attention;
// GEMM A-tiles staged into LDS via the Tensor Data Mover (tensor_load_to_lds).
// ---------------------------------------------------------------------------

typedef __attribute__((ext_vector_type(16))) _Float16 v16h;
typedef __attribute__((ext_vector_type(8)))  _Float16 v8h;
typedef __attribute__((ext_vector_type(8)))  float    v8f;
typedef __attribute__((ext_vector_type(4)))  unsigned int v4u;
typedef __attribute__((ext_vector_type(8)))  int      v8i;
typedef __attribute__((ext_vector_type(4)))  int      v4i;

constexpr int B_ = 2, T_ = 2048, S_ = 2048, C_ = 1024, H_ = 16, D_ = 64;
constexpr int BT_ = B_ * T_;   // 4096 rows for activation GEMMs

#define DEV static __device__ __forceinline__

DEV v8f wmma_f16(v16h a, v16h b, v8f c) {
  // 8 args: (neg_a, A, neg_b, B, c_mod, C, reuse_a, reuse_b)
  return __builtin_amdgcn_wmma_f32_16x16x32_f16(false, a, false, b, (short)0, c,
                                                false, false);
}

DEV v16h combine8(v8h lo, v8h hi) {
  v16h r;
#pragma unroll
  for (int j = 0; j < 8; ++j) { r[j] = lo[j]; r[j + 8] = hi[j]; }
  return r;
}

// A-matrix fragment (16x32 f16): lanes 0-15 hold row=lane,
// K = 0..15; lanes 16-31 hold row=lane-16, K = 16..31.
DEV v16h load_frag_a(const _Float16* __restrict__ tile, int stride, int lane) {
  const int row  = lane & 15;
  const int koff = (lane >> 4) * 16;
  const _Float16* p = tile + row * stride + koff;
  return combine8(*(const v8h*)p, *(const v8h*)(p + 8));
}

// B-matrix fragment (32x16 f16): lanes 0-15 hold col=lane, K=0..15;
// lanes 16-31 hold col=lane-16, K=16..31.  With B stored col-major
// (stride between columns) each lane reads 16 contiguous halves.
DEV v16h load_frag_b_cm(const _Float16* __restrict__ tileT, int stride, int lane) {
  const int col  = lane & 15;
  const int koff = (lane >> 4) * 16;
  return *(const v16h*)(tileT + col * stride + koff);
}

// ---------------------------------------------------------------------------
// Tensor Data Mover: DMA a 2-D f16 tile [rows x cols] (row stride
// row_stride_elems) from global memory into LDS at lds_off, packed
// contiguously as [rows][cols].  D# layout per CDNA5 ISA
// §8.3/8.4; groups 2/3 zero (<=2-D tile).  Tracked by TENSORcnt.
// This toolchain exposes the 6-arg builtin:
//   (uint32x4 g0, int32x8 g1, int32x4 g2, int32x4 g3, int32x8 extra, i32 cpol)
// ---------------------------------------------------------------------------
DEV void tdm_load_tile_2d(const _Float16* gptr, unsigned lds_off,
                          int rows, int cols, int row_stride_elems,
                          int tensor_d0, int tensor_d1) {
  unsigned long long ga = (unsigned long long)(uintptr_t)gptr;
  v4u g0;
  g0[0] = 1u;                                    // count=1, user descriptor
  g0[1] = lds_off;                               // lds_addr (bytes)
  g0[2] = (unsigned)(ga & 0xffffffffu);          // global_addr[31:0]
  g0[3] = (unsigned)((ga >> 32) & 0x01ffffffu)   // global_addr[56:32]
          | 0x80000000u;                         // type=2 ("image")
  v8i g1;
  g1[0] = 0x00010000;                            // data_size=1 (2 bytes)
  g1[1] = (int)(((unsigned)tensor_d0 & 0xffffu) << 16);   // dim0[15:0] @63:48
  g1[2] = (int)((((unsigned)tensor_d0 >> 16) & 0xffffu)   // dim0[31:16]
          | (((unsigned)tensor_d1 & 0xffffu) << 16));     // dim1[15:0]
  g1[3] = (int)((((unsigned)tensor_d1 >> 16) & 0xffffu)   // dim1[31:16]
          | (((unsigned)cols & 0xffffu) << 16));          // tile_dim0
  g1[4] = rows & 0xffff;                         // tile_dim1 (tile_dim2=0)
  g1[5] = row_stride_elems;                      // dim0_stride[31:0]
  g1[6] = 0;                                     // dim0_stride[47:32]
  g1[7] = 0;
  v4i z4 = {0, 0, 0, 0};
  v8i z8 = {0, 0, 0, 0, 0, 0, 0, 0};
  __builtin_amdgcn_tensor_load_to_lds(g0, g1, z4, z4, z8, /*cpol=*/0);
}

// ---------------------------------------------------------------------------
// fp32 -> f16 cast, 4 elements/thread
// ---------------------------------------------------------------------------
struct H4 { _Float16 x, y, z, w; };

__global__ void cast_f32_to_f16_v4(const float4* __restrict__ in,
                                   H4* __restrict__ out, int n4) {
  int i = blockIdx.x * blockDim.x + threadIdx.x;
  if (i < n4) {
    float4 v = in[i];
    out[i] = H4{(_Float16)v.x, (_Float16)v.y, (_Float16)v.z, (_Float16)v.w};
  }
}

// ---------------------------------------------------------------------------
// GEMM: C[M,N] = A[M,K] * W[K,N], f16 inputs, fp32 accumulate.
// BM=128 BN=64 BK=64, 256 threads = 8 waves, each wave a 32x32 tile.
// A tile DMA'd into LDS by the TDM (wave 0); W tile staged transposed by all.
// ---------------------------------------------------------------------------
template <bool OUT_F32>
__global__ __launch_bounds__(256)
void gemm_f16_wmma(const _Float16* __restrict__ A,
                   const _Float16* __restrict__ W,
                   void* __restrict__ Cout, int M, int N, int K) {
  constexpr int BM = 128, BN = 64, BK = 64;
  __shared__ _Float16 As[BM * BK];     // row-major [m][k], filled by TDM
  __shared__ _Float16 Bt[BN * BK];     // transposed [n][k]

  const int tid  = threadIdx.x;
  const int lane = tid & 31;
  const int wave = tid >> 5;
  const int m0 = blockIdx.y * BM;
  const int n0 = blockIdx.x * BN;
  const int wm = (wave >> 1) * 32;     // wave M offset in block tile
  const int wn = (wave & 1) * 32;      // wave N offset in block tile
  const int rowsel = lane & 15;
  const unsigned as_lds = (unsigned)(uintptr_t)(void*)&As[0];

  v8f acc[2][2] = {};

  for (int k0 = 0; k0 < K; k0 += BK) {
    // A tile: one TDM descriptor moves 128x64 f16 while we stage W
    if (wave == 0)
      tdm_load_tile_2d(&A[(size_t)m0 * K + k0], as_lds, BM, BK, K, K, M);

    // stage W tile transposed: 64x64 halves = 512 chunks, 2 per thread
#pragma unroll
    for (int c = 0; c < 2; ++c) {
      int chunk = tid + c * 256;
      int kr = chunk >> 3;
      int nc = (chunk & 7) * 8;
      v8h v = *(const v8h*)(&W[(k0 + kr) * N + n0 + nc]);
#pragma unroll
      for (int j = 0; j < 8; ++j) Bt[(nc + j) * BK + kr] = v[j];
    }
    // prefetch next W tile into L2 while we compute
    if (k0 + BK < K)
      __builtin_prefetch(&W[(k0 + BK + (tid >> 2)) * N + n0], 0, 1);

    if (wave == 0) __builtin_amdgcn_s_wait_tensorcnt(0);
    __syncthreads();

#pragma unroll
    for (int ks = 0; ks < BK; ks += 32) {
      v16h af[2], bf[2];
#pragma unroll
      for (int mi = 0; mi < 2; ++mi)
        af[mi] = load_frag_a(&As[(wm + mi * 16) * BK + ks], BK, lane);
#pragma unroll
      for (int ni = 0; ni < 2; ++ni)
        bf[ni] = load_frag_b_cm(&Bt[(wn + ni * 16) * BK + ks], BK, lane);
#pragma unroll
      for (int mi = 0; mi < 2; ++mi)
#pragma unroll
        for (int ni = 0; ni < 2; ++ni)
          acc[mi][ni] = wmma_f16(af[mi], bf[ni], acc[mi][ni]);
    }
    __syncthreads();
  }

  // C/D layout: lanes 0-15 -> rows j (VGPR j), N=lane; lanes 16-31 -> rows j+8
  const int hi = lane >> 4;
#pragma unroll
  for (int mi = 0; mi < 2; ++mi)
#pragma unroll
    for (int ni = 0; ni < 2; ++ni)
#pragma unroll
      for (int j = 0; j < 8; ++j) {
        int row = m0 + wm + mi * 16 + hi * 8 + j;
        int col = n0 + wn + ni * 16 + rowsel;
        if (OUT_F32)
          ((float*)Cout)[(size_t)row * N + col] = acc[mi][ni][j];
        else
          ((_Float16*)Cout)[(size_t)row * N + col] = (_Float16)acc[mi][ni][j];
      }
}

// ---------------------------------------------------------------------------
// Flash attention: one wave owns a 16-row Q tile for one (b,h); streams S in
// chunks of 32 with online softmax.  4 waves/block, disjoint LDS slices.
// ---------------------------------------------------------------------------
__global__ __launch_bounds__(128)
void attn_flash_wmma(const _Float16* __restrict__ Qh,
                     const _Float16* __restrict__ Kh,
                     const _Float16* __restrict__ Vh,
                     _Float16* __restrict__ Yatt) {
  __shared__ _Float16 Pl[4][16 * 32];   // per-wave prob tile (C-layout -> A-layout bounce)

  const int lane   = threadIdx.x & 31;
  const int wave   = threadIdx.x >> 5;
  const int b      = blockIdx.z;
  const int h      = blockIdx.y;
  const int t0     = (blockIdx.x * 4 + wave) * 16;
  const int rowsel = lane & 15;
  const int hi     = lane >> 4;
  _Float16* P = &Pl[wave][0];

  const _Float16* Qb = Qh + (size_t)(b * T_ + t0) * C_ + h * D_;
  const _Float16* Kb = Kh + (size_t)(b * S_) * C_ + h * D_;
  const _Float16* Vb = Vh + (size_t)(b * S_) * C_ + h * D_;

  // Q fragments for D chunks k=0..31 and k=32..63 (held for whole S loop)
  v16h qf[2];
#pragma unroll
  for (int kk = 0; kk < 2; ++kk)
    qf[kk] = load_frag_a(Qb + kk * 32, C_, lane);

  float m[8], l[8];
#pragma unroll
  for (int j = 0; j < 8; ++j) { m[j] = -3.0e38f; l[j] = 0.f; }
  v8f acc[4] = {};                      // 16 x 64 output accumulator

  const float scale = 0.125f;           // 1/sqrt(64)

  for (int s0 = 0; s0 < S_; s0 += 32) {
    // ---- scores: Q[16,64] @ K[s0..s0+31, 64]^T, two 16-col tiles ----
    v8f sc0 = {}, sc1 = {};
#pragma unroll
    for (int kk = 0; kk < 2; ++kk) {
      // B[k][n] = K[s0+n][kk*32+k] -> contiguous 16 halves per lane
      v16h b0 = *(const v16h*)(Kb + (size_t)(s0 + rowsel) * C_ + kk * 32 + hi * 16);
      v16h b1 = *(const v16h*)(Kb + (size_t)(s0 + 16 + rowsel) * C_ + kk * 32 + hi * 16);
      sc0 = wmma_f16(qf[kk], b0, sc0);
      sc1 = wmma_f16(qf[kk], b1, sc1);
    }

    // ---- online softmax over the 32 new columns ----
    float corr[8];
#pragma unroll
    for (int j = 0; j < 8; ++j) {
      float x0 = sc0[j] * scale, x1 = sc1[j] * scale;
      float mt = fmaxf(x0, x1);                // row spans 16 lanes of this half
      mt = fmaxf(mt, __shfl_xor(mt, 1));
      mt = fmaxf(mt, __shfl_xor(mt, 2));
      mt = fmaxf(mt, __shfl_xor(mt, 4));
      mt = fmaxf(mt, __shfl_xor(mt, 8));
      float mn = fmaxf(m[j], mt);
      float cf = __expf(m[j] - mn);
      float p0 = __expf(x0 - mn);
      float p1 = __expf(x1 - mn);
      float rs = p0 + p1;
      rs += __shfl_xor(rs, 1);
      rs += __shfl_xor(rs, 2);
      rs += __shfl_xor(rs, 4);
      rs += __shfl_xor(rs, 8);
      l[j] = l[j] * cf + rs;
      m[j] = mn;
      corr[j] = cf;
      // stash probs (C layout: row = j + hi*8, col = rowsel) into LDS
      P[(j + hi * 8) * 32 + rowsel]      = (_Float16)p0;
      P[(j + hi * 8) * 32 + 16 + rowsel] = (_Float16)p1;
    }
#pragma unroll
    for (int n = 0; n < 4; ++n)
#pragma unroll
      for (int j = 0; j < 8; ++j) acc[n][j] *= corr[j];

    __builtin_amdgcn_wave_barrier();            // order DS stores before loads
    v16h pf = load_frag_a(P, 32, lane);         // 16x32 A-fragment of probs
    __builtin_amdgcn_wave_barrier();

    // ---- acc += P[16,32] @ V[s0..s0+31, 0..63] ----
#pragma unroll
    for (int n = 0; n < 4; ++n) {
      // B[k][nn] = V[s0 + koff + k][n*16 + col], koff = hi*16
      const _Float16* vp = Vb + (size_t)(s0 + hi * 16) * C_ + n * 16 + rowsel;
      v16h bf;
#pragma unroll
      for (int k = 0; k < 16; ++k) bf[k] = vp[(size_t)k * C_];
      acc[n] = wmma_f16(pf, bf, acc[n]);
    }
  }

  // ---- finalize: divide by row sums, store f16 ----
#pragma unroll
  for (int j = 0; j < 8; ++j) l[j] = 1.0f / l[j];
  _Float16* Ob = Yatt + (size_t)(b * T_ + t0) * C_ + h * D_;
#pragma unroll
  for (int n = 0; n < 4; ++n)
#pragma unroll
    for (int j = 0; j < 8; ++j)
      Ob[(size_t)(j + hi * 8) * C_ + n * 16 + rowsel] = (_Float16)(acc[n][j] * l[j]);
}

// ---------------------------------------------------------------------------
// Launch
// ---------------------------------------------------------------------------
extern "C" void kernel_launch(void* const* d_in, const int* in_sizes, int n_in,
                              void* d_out, int out_size, void* d_ws, size_t ws_size,
                              hipStream_t stream) {
  (void)in_sizes; (void)n_in; (void)out_size; (void)ws_size;
  const float* q  = (const float*)d_in[0];
  const float* p  = (const float*)d_in[1];
  const float* Wq = (const float*)d_in[2];
  const float* Wk = (const float*)d_in[3];
  const float* Wv = (const float*)d_in[4];
  const float* Wo = (const float*)d_in[5];
  float* out = (float*)d_out;

  const size_t nAct = (size_t)B_ * T_ * C_;   // 4,194,304
  const size_t nW   = (size_t)C_ * C_;        // 1,048,576
  _Float16* ws   = (_Float16*)d_ws;           // ~56 MB of f16 scratch total
  _Float16* q16  = ws;
  _Float16* p16  = q16 + nAct;
  _Float16* Wq16 = p16 + nAct;
  _Float16* Wk16 = Wq16 + nW;
  _Float16* Wv16 = Wk16 + nW;
  _Float16* Wo16 = Wv16 + nW;
  _Float16* Qh   = Wo16 + nW;
  _Float16* Kh   = Qh + nAct;
  _Float16* Vh   = Kh + nAct;
  _Float16* Ya   = Vh + nAct;

  const int actBlocks = (int)((nAct / 4 + 255) / 256);
  const int wBlocks   = (int)((nW / 4 + 255) / 256);
  cast_f32_to_f16_v4<<<actBlocks, 256, 0, stream>>>((const float4*)q,  (H4*)q16,  (int)(nAct / 4));
  cast_f32_to_f16_v4<<<actBlocks, 256, 0, stream>>>((const float4*)p,  (H4*)p16,  (int)(nAct / 4));
  cast_f32_to_f16_v4<<<wBlocks,   256, 0, stream>>>((const float4*)Wq, (H4*)Wq16, (int)(nW / 4));
  cast_f32_to_f16_v4<<<wBlocks,   256, 0, stream>>>((const float4*)Wk, (H4*)Wk16, (int)(nW / 4));
  cast_f32_to_f16_v4<<<wBlocks,   256, 0, stream>>>((const float4*)Wv, (H4*)Wv16, (int)(nW / 4));
  cast_f32_to_f16_v4<<<wBlocks,   256, 0, stream>>>((const float4*)Wo, (H4*)Wo16, (int)(nW / 4));

  dim3 gg(C_ / 64, BT_ / 128);                 // 16 x 32 blocks
  gemm_f16_wmma<false><<<gg, 256, 0, stream>>>(q16, Wq16, Qh, BT_, C_, C_);
  gemm_f16_wmma<false><<<gg, 256, 0, stream>>>(p16, Wk16, Kh, BT_, C_, C_);
  gemm_f16_wmma<false><<<gg, 256, 0, stream>>>(p16, Wv16, Vh, BT_, C_, C_);

  attn_flash_wmma<<<dim3(T_ / 64, H_, B_), 128, 0, stream>>>(Qh, Kh, Vh, Ya);

  gemm_f16_wmma<true><<<gg, 256, 0, stream>>>(Ya, Wo16, out, BT_, C_, C_);
}
